// ChemicalConvWithBonds_9131100472088
// MI455X (gfx1250) — matrix operand order repaired and checked by reading, not compiled, not run on hardware
//
#include <hip/hip_runtime.h>

typedef __attribute__((ext_vector_type(16))) __bf16 v16bf;
typedef __attribute__((ext_vector_type(8)))  float  v8f;

#define HID 128
#define W1S 296                         // padded K-stride (bf16) of W1^T rows (288 used)
#define W2S 136                         // padded K-stride (bf16) of W2^T rows
#define W1T_USHORTS (128 * W1S)         // 37888
#define W2T_USHORTS (128 * W2S)         // 17408
#define WT_USHORTS  (W1T_USHORTS + W2T_USHORTS)   // 55296
#define SY_USHORTS  (8 * 16 * HID)      // 16384
#define SMEM_USHORTS (WT_USHORTS + SY_USHORTS)
#define HB_OFF_BYTES (WT_USHORTS * 2)   // 110592 (16B aligned)
#define TPW 4                           // edge-tiles per wave
#define EPB (8 * 16 * TPW)              // 512 edges per block

__device__ __forceinline__ unsigned short f2bf(float f) {
  union { float f; unsigned u; } a{f};
  return (unsigned short)((a.u + 0x8000u) >> 16);
}
__device__ __forceinline__ unsigned f2bf2(float lo, float hi) {
  return (unsigned)f2bf(lo) | ((unsigned)f2bf(hi) << 16);
}
__device__ __forceinline__ v16bf packA(float4 f0, float4 f1, float4 f2, float4 f3) {
  v16bf a;
  unsigned* u = reinterpret_cast<unsigned*>(&a);
  u[0]=f2bf2(f0.x,f0.y); u[1]=f2bf2(f0.z,f0.w);
  u[2]=f2bf2(f1.x,f1.y); u[3]=f2bf2(f1.z,f1.w);
  u[4]=f2bf2(f2.x,f2.y); u[5]=f2bf2(f2.z,f2.w);
  u[6]=f2bf2(f3.x,f3.y); u[7]=f2bf2(f3.z,f3.w);
  return a;
}
__device__ __forceinline__ v16bf ldB(const unsigned short* s) {   // chunks +0, +16
  v16bf r;
  uint4* d = reinterpret_cast<uint4*>(&r);
  d[0] = *reinterpret_cast<const uint4*>(s);
  d[1] = *reinterpret_cast<const uint4*>(s + 16);
  return r;
}
__device__ __forceinline__ v16bf ldA16(const unsigned short* s) { // chunks +0, +16
  v16bf r;
  uint4* d = reinterpret_cast<uint4*>(&r);
  d[0] = *reinterpret_cast<const uint4*>(s);
  d[1] = *reinterpret_cast<const uint4*>(s + 16);
  return r;
}
__device__ __forceinline__ v8f wmma_bf16(v16bf a, v16bf b, v8f c) {
  return __builtin_amdgcn_wmma_f32_16x16x32_bf16(false, a, false, b, (short)0, c, false, false);
}

// ---------------- precompute kernels ----------------
__global__ void zero_f4(float4* __restrict__ p, int n4) {
  int i = blockIdx.x * blockDim.x + threadIdx.x;
  if (i < n4) p[i] = make_float4(0.f, 0.f, 0.f, 0.f);
}
__global__ void conv_h(const float2* __restrict__ h2, unsigned* __restrict__ o, int nPairs) {
  int i = blockIdx.x * blockDim.x + threadIdx.x;
  if (i < nPairs) { float2 v = h2[i]; o[i] = f2bf2(v.x, v.y); }
}
__global__ void conv_w(const float* __restrict__ W1, const float* __restrict__ W2,
                       unsigned short* __restrict__ wt) {
  int i = blockIdx.x * blockDim.x + threadIdx.x;
  if (i < 272 * 128) {
    int k = i >> 7, n = i & 127;
    wt[n * W1S + k] = f2bf(W1[i]);
  } else if (i < 272 * 128 + 128 * 24) {
    int j = i - 272 * 128;
    int n = j / 24, k = 272 + j % 24;
    wt[n * W1S + k] = 0;
  } else if (i < 272 * 128 + 128 * 24 + 128 * 128) {
    int j = i - 272 * 128 - 128 * 24;
    int k = j >> 7, n = j & 127;
    wt[W1T_USHORTS + n * W2S + k] = f2bf(W2[j]);
  }
}
#define CONVW_THREADS (272*128 + 128*24 + 128*128)

// ---------------- main kernel (FAST: pre-converted bf16; else inline convert) ----------------
template <bool FAST>
__global__ void __launch_bounds__(256)
mpnn_main(const unsigned short* __restrict__ hb, const float* __restrict__ hf,
          const int* __restrict__ ei, const float* __restrict__ bond,
          const unsigned short* __restrict__ wt,
          const float* __restrict__ W1, const float* __restrict__ W2,
          const float* __restrict__ b1, const float* __restrict__ b2,
          float* __restrict__ out, int E)
{
  extern __shared__ __align__(16) unsigned short smem[];
  unsigned short* sW1 = smem;
  unsigned short* sW2 = smem + W1T_USHORTS;
  unsigned short* sY  = smem + WT_USHORTS;
  const int tid = threadIdx.x;

  if constexpr (FAST) {      // bulk b128 copy of pre-transposed bf16 weights
    const uint4* s4 = (const uint4*)wt;
    uint4* d4 = (uint4*)smem;
    #pragma unroll 1
    for (int i = tid; i < (WT_USHORTS * 2) / 16; i += 256) d4[i] = s4[i];
  } else {                   // convert+transpose from f32 in global
    for (int i = tid; i < 272 * 128; i += 256) {
      int k = i >> 7, n = i & 127;
      sW1[n * W1S + k] = f2bf(W1[i]);
    }
    for (int i = tid; i < 128 * 24; i += 256) {
      int n = i / 24, k = 272 + i % 24;
      sW1[n * W1S + k] = 0;
    }
    for (int i = tid; i < 128 * 128; i += 256) {
      int k = i >> 7, n = i & 127;
      sW2[n * W2S + k] = f2bf(W2[i]);
    }
  }
  __syncthreads();

  const int wave = tid >> 5, lane = tid & 31;
  const int r    = lane & 15;
  const int hi   = lane >> 4;
  const int aOff = hi * 8;
  const int bOff = hi * 8;
  unsigned short* y = sY + wave * (16 * HID);

  float bias1[8], bias2[8];                 // loop-invariant per-lane biases
  #pragma unroll
  for (int n = 0; n < 8; ++n) { bias1[n] = b1[n * 16 + r]; bias2[n] = b2[n * 16 + r]; }

  const long waveBase = (long)blockIdx.x * EPB + wave * (16 * TPW);

  #pragma unroll 1
  for (int t = 0; t < TPW; ++t) {
    const long eBase = waveBase + t * 16;
    const long e  = eBase + r;
    const long ec = (e < E) ? e : (long)(E - 1);
    const int src = ei[ec];
    const int dst = ei[(long)E + ec];

    v8f acc[8];
    #pragma unroll
    for (int n = 0; n < 8; ++n)
      acc[n] = (v8f){0.f,0.f,0.f,0.f,0.f,0.f,0.f,0.f};

    #pragma unroll
    for (int half = 0; half < 2; ++half) {               // K tiles 0-7: h[row], h[col]
      const int node = half ? dst : src;
      #pragma unroll
      for (int kt = 0; kt < 4; ++kt) {
        v16bf a;
        if constexpr (FAST) {
          a = ldA16(hb + (size_t)node * HID + kt * 32 + aOff);
        } else {
          const float* p = hf + (size_t)node * HID + kt * 32 + aOff;
          a = packA(*(const float4*)(p),      *(const float4*)(p + 4),
                    *(const float4*)(p + 16), *(const float4*)(p + 20));
        }
        v16bf bfr[8];
        #pragma unroll
        for (int n = 0; n < 8; ++n)
          bfr[n] = ldB(sW1 + (n * 16 + r) * W1S + half * 128 + kt * 32 + bOff);
        __builtin_amdgcn_sched_barrier(0);               // batch loads before WMMA chain
        #pragma unroll
        for (int n = 0; n < 8; ++n) acc[n] = wmma_bf16(a, bfr[n], acc[n]);
      }
    }
    {                                                    // K tile 8: bond (16 + 16 zero-pad)
      const float* p = bond + ec * 16 + aOff;
      float4 z = make_float4(0.f, 0.f, 0.f, 0.f);
      v16bf a = packA(*(const float4*)(p), *(const float4*)(p + 4), z, z);
      v16bf bfr[8];
      #pragma unroll
      for (int n = 0; n < 8; ++n)
        bfr[n] = ldB(sW1 + (n * 16 + r) * W1S + 256 + bOff);
      __builtin_amdgcn_sched_barrier(0);
      #pragma unroll
      for (int n = 0; n < 8; ++n) acc[n] = wmma_bf16(a, bfr[n], acc[n]);
    }

    // bias + SiLU, write bf16 Y tile (C-layout -> row-major, intra-wave transpose via LDS)
    #pragma unroll
    for (int n = 0; n < 8; ++n) {
      #pragma unroll
      for (int j = 0; j < 8; ++j) {
        float zv = acc[n][j] + bias1[n];
        float sv = zv / (1.f + __expf(-zv));             // SiLU
        y[(j + hi * 8) * HID + n * 16 + r] = f2bf(sv);
      }
    }
    __syncthreads();

    v8f acc2[8];
    #pragma unroll
    for (int n = 0; n < 8; ++n)
      acc2[n] = (v8f){0.f,0.f,0.f,0.f,0.f,0.f,0.f,0.f};
    #pragma unroll
    for (int kt = 0; kt < 4; ++kt) {                     // layer 2
      v16bf a = ldA16(y + r * HID + kt * 32 + aOff);
      v16bf bfr[8];
      #pragma unroll
      for (int n = 0; n < 8; ++n)
        bfr[n] = ldB(sW2 + (n * 16 + r) * W2S + kt * 32 + bOff);
      __builtin_amdgcn_sched_barrier(0);
      #pragma unroll
      for (int n = 0; n < 8; ++n) acc2[n] = wmma_bf16(a, bfr[n], acc2[n]);
    }

    #pragma unroll
    for (int j = 0; j < 8; ++j) {                        // scatter-add to out[row]
      const long em = eBase + j + hi * 8;
      if (em < E) {
        const int srcm = ei[em];
        float* orow = out + (size_t)srcm * HID;
        #pragma unroll
        for (int n = 0; n < 8; ++n)
          __hip_atomic_fetch_add(orow + n * 16 + r, acc2[n][j] + bias2[n],
                                 __ATOMIC_RELAXED, __HIP_MEMORY_SCOPE_AGENT);
      }
    }
  }
}

extern "C" void kernel_launch(void* const* d_in, const int* in_sizes, int n_in,
                              void* d_out, int out_size, void* d_ws, size_t ws_size,
                              hipStream_t stream) {
  const float* h    = (const float*)d_in[0];
  const int*   ei   = (const int*)  d_in[1];
  const float* bond = (const float*)d_in[2];
  const float* W1   = (const float*)d_in[3];
  const float* b1   = (const float*)d_in[4];
  const float* W2   = (const float*)d_in[5];
  const float* b2   = (const float*)d_in[6];
  float* out = (float*)d_out;

  const int E      = in_sizes[1] / 2;
  const int nNodes = in_sizes[0] / HID;

  const int n4 = out_size / 4;
  zero_f4<<<(n4 + 255) / 256, 256, 0, stream>>>((float4*)out, n4);

  const int nBlocks   = (E + EPB - 1) / EPB;
  const size_t smem   = (size_t)SMEM_USHORTS * sizeof(unsigned short);
  const size_t wsNeed = (size_t)HB_OFF_BYTES + (size_t)nNodes * HID * 2;

  if (ws_size >= wsNeed) {
    unsigned short* wt  = (unsigned short*)d_ws;
    unsigned*       hbu = (unsigned*)((char*)d_ws + HB_OFF_BYTES);
    conv_w<<<(CONVW_THREADS + 255) / 256, 256, 0, stream>>>(W1, W2, wt);
    const int nPairs = nNodes * HID / 2;
    conv_h<<<(nPairs + 255) / 256, 256, 0, stream>>>((const float2*)h, hbu, nPairs);
    mpnn_main<true><<<nBlocks, 256, smem, stream>>>(
        (const unsigned short*)hbu, nullptr, ei, bond, wt,
        nullptr, nullptr, b1, b2, out, E);
  } else {
    mpnn_main<false><<<nBlocks, 256, smem, stream>>>(
        nullptr, h, ei, bond, nullptr,
        W1, W2, b1, b2, out, E);
  }
}
